// LGCEncoder_72164040508243
// MI455X (gfx1250) — compile-verified, matrix operand order checked by reading, not run on hardware
//
#include <hip/hip_runtime.h>

typedef _Float16 v4h  __attribute__((ext_vector_type(4)));
typedef _Float16 v8h  __attribute__((ext_vector_type(8)));
typedef _Float16 v16h __attribute__((ext_vector_type(16)));
typedef float    v8f  __attribute__((ext_vector_type(8)));

#define B_TOTAL   8192
#define NB        64
#define FDIM      256
#define HDIM      256
#define ODIM      128
#define KS        5
#define TOPK      8
#define XROWS     9            // 1 node row + 8 top-k rows
#define KDIM      (KS * 256)   // 1280 (same for conv1 and conv2: 5 taps x 256 ch)
#define SPB       16           // samples per block
#define XPITCH    264          // f16 pitch (256 + 8 pad) to spread LDS banks
#define HPITCH    264

#define XS_BYTES  (SPB * XROWS * XPITCH * 2)   // 76032
#define HS_BYTES  (SPB * KS * HPITCH * 2)      // 42240
#define SMEM_BYTES (XS_BYTES + HS_BYTES)       // 118272

static __device__ inline v16h cat16(v8h lo, v8h hi) {
  return __builtin_shufflevector(lo, hi, 0,1,2,3,4,5,6,7,8,9,10,11,12,13,14,15);
}

// ---------------------------------------------------------------------------
// Prep: transpose W1 [1280,256] -> w1t[n][k] f16, W2 [1280,128] -> w2t[n][k].
// K index k = tap*256 + in_channel, matching the im2col A-row layout, so
// B-matrix fragments become two contiguous 16B loads per lane.
// ---------------------------------------------------------------------------
__global__ void lgc_prep_weights(const float* __restrict__ W1,
                                 const float* __restrict__ W2,
                                 _Float16* __restrict__ w1t,
                                 _Float16* __restrict__ w2t) {
  int idx = blockIdx.x * blockDim.x + threadIdx.x;
  const int n1 = HDIM * KDIM;   // 327680
  const int n2 = ODIM * KDIM;   // 163840
  if (idx < n1) {
    int n = idx / KDIM, k = idx % KDIM;
    w1t[idx] = (_Float16)W1[(size_t)k * HDIM + n];
  } else if (idx < n1 + n2) {
    int i = idx - n1;
    int n = i / KDIM, k = i % KDIM;
    w2t[i] = (_Float16)W2[(size_t)k * ODIM + n];
  }
}

// ---------------------------------------------------------------------------
// Fused main kernel: gather + per-channel top-8 + conv1 + conv2 via WMMA f16.
// One workgroup = 16 samples, 256 threads = 8 waves (wave32).
// ---------------------------------------------------------------------------
__global__ __launch_bounds__(256) void lgc_main(
    const int*      __restrict__ node_ids,
    const int*      __restrict__ neighbors,
    const float*    __restrict__ feat,
    const _Float16* __restrict__ w1t,
    const float*    __restrict__ b1,
    const _Float16* __restrict__ w2t,
    const float*    __restrict__ b2,
    float*          __restrict__ out)
{
  extern __shared__ char smem[];
  _Float16* xs = (_Float16*)(smem);             // [SPB][XROWS][XPITCH]
  _Float16* hs = (_Float16*)(smem + XS_BYTES);  // [SPB][KS][HPITCH]

  const int tid   = threadIdx.x;
  const int lane  = tid & 31;
  const int wv    = tid >> 5;         // 0..7
  const int sb    = blockIdx.x * SPB; // first sample of this block

  // ========================= Phase 1: gather + top-8 =======================
  // Thread t owns 4 channels (ch4 = t&63 -> channels 4*ch4..4*ch4+3) of 4
  // samples (sgroup = t>>6). Neighbor-row reads are 512B-contiguous per wave.
  {
    const int ch4 = tid & 63;
    const int sg  = tid >> 6;       // 0..3
    for (int rep = 0; rep < 4; ++rep) {
      const int s = sg * 4 + rep;
      const int* nrow = neighbors + (size_t)(sb + s) * NB;

      float t0[8], t1[8], t2[8], t3[8];
      #pragma unroll
      for (int i = 0; i < 8; ++i)
        t0[i] = t1[i] = t2[i] = t3[i] = -__builtin_inff();

      for (int j = 0; j < NB; ++j) {
        size_t row = (size_t)nrow[j] * FDIM;
        float4 v = ((const float4*)(feat + row))[ch4];
        float v0 = v.x, v1 = v.y, v2 = v.z, v3 = v.w;
        if (v0 > t0[7]) {
          #pragma unroll
          for (int i = 0; i < 8; ++i) { float hi = fmaxf(t0[i], v0), lo = fminf(t0[i], v0); t0[i] = hi; v0 = lo; }
        }
        if (v1 > t1[7]) {
          #pragma unroll
          for (int i = 0; i < 8; ++i) { float hi = fmaxf(t1[i], v1), lo = fminf(t1[i], v1); t1[i] = hi; v1 = lo; }
        }
        if (v2 > t2[7]) {
          #pragma unroll
          for (int i = 0; i < 8; ++i) { float hi = fmaxf(t2[i], v2), lo = fminf(t2[i], v2); t2[i] = hi; v2 = lo; }
        }
        if (v3 > t3[7]) {
          #pragma unroll
          for (int i = 0; i < 8; ++i) { float hi = fmaxf(t3[i], v3), lo = fminf(t3[i], v3); t3[i] = hi; v3 = lo; }
        }
      }

      // node feature -> x row 0
      {
        size_t row = (size_t)node_ids[sb + s] * FDIM;
        float4 v = ((const float4*)(feat + row))[ch4];
        v4h p; p.x = (_Float16)v.x; p.y = (_Float16)v.y;
        p.z = (_Float16)v.z; p.w = (_Float16)v.w;
        *(v4h*)(xs + (size_t)(s * XROWS + 0) * XPITCH + ch4 * 4) = p;
      }
      // top-8 (descending) -> x rows 1..8
      #pragma unroll
      for (int r = 0; r < TOPK; ++r) {
        v4h p; p.x = (_Float16)t0[r]; p.y = (_Float16)t1[r];
        p.z = (_Float16)t2[r]; p.w = (_Float16)t3[r];
        *(v4h*)(xs + (size_t)(s * XROWS + 1 + r) * XPITCH + ch4 * 4) = p;
      }
    }
  }
  __syncthreads();

  // ========================= Phase 2: conv1 (WMMA) =========================
  // GEMM: M = 16 samples * 5 positions = 80 rows (5 tiles of 16),
  //       N = 256 (16 tiles), K = 1280 (40 steps of 32, each inside one tap).
  const int nlane = lane & 15;
  const int khalf = lane >> 4;

  int abase[5];
  #pragma unroll
  for (int mt = 0; mt < 5; ++mt) {
    int r  = mt * 16 + nlane;     // global A row 0..79
    int bl = r / 5, p = r % 5;    // sample-local, output position
    abase[mt] = bl * XROWS + p;   // x row base; + tap at each K step
  }

  #pragma unroll
  for (int nn = 0; nn < 2; ++nn) {
    const int nt = wv * 2 + nn;            // N tile 0..15
    const int ng = nt * 16 + nlane;        // output channel 0..255
    const _Float16* bptr = w1t + (size_t)ng * KDIM;

    v8f acc[5];
    #pragma unroll
    for (int mt = 0; mt < 5; ++mt) { v8f z = {}; acc[mt] = z; }

    for (int step = 0; step < 40; ++step) {
      const int kb = step * 32;
      // B fragment: two contiguous groups of 8 f16 (documented lane/K layout)
      v8h blo = *(const v8h*)(bptr + kb + 8 * khalf);
      v8h bhi = *(const v8h*)(bptr + kb + 16 + 8 * khalf);
      v16h bf = cat16(blo, bhi);

      const int tap = step >> 3;
      const int cb  = (step & 7) * 32;
      const int c1  = cb + 8 * khalf;
      const int c2  = cb + 16 + 8 * khalf;
      #pragma unroll
      for (int mt = 0; mt < 5; ++mt) {
        const _Float16* ap = xs + (size_t)(abase[mt] + tap) * XPITCH;
        v16h af = cat16(*(const v8h*)(ap + c1), *(const v8h*)(ap + c2));
        acc[mt] = __builtin_amdgcn_wmma_f32_16x16x32_f16(
            false, af, false, bf, (short)0, acc[mt], false, false);
      }
    }

    const float bias = b1[ng];
    #pragma unroll
    for (int mt = 0; mt < 5; ++mt) {
      #pragma unroll
      for (int v = 0; v < 8; ++v) {
        int m  = v + 8 * khalf;          // D layout: VGPR v, lane half
        int r  = mt * 16 + m;
        int bl = r / 5, p = r % 5;
        hs[(size_t)(bl * KS + p) * HPITCH + ng] = (_Float16)(acc[mt][v] + bias);
      }
    }
  }
  __syncthreads();

  // ========================= Phase 3: conv2 (WMMA) =========================
  // GEMM: M = 16 samples (1 tile), N = 128 (8 tiles, one per wave), K = 1280.
  {
    const int ng = wv * 16 + nlane;      // output channel 0..127
    const _Float16* bptr = w2t + (size_t)ng * KDIM;
    const int m = nlane;                 // sample-local row
    v8f acc = {};

    for (int step = 0; step < 40; ++step) {
      const int kb = step * 32;
      v16h bf = cat16(*(const v8h*)(bptr + kb + 8 * khalf),
                      *(const v8h*)(bptr + kb + 16 + 8 * khalf));
      const int tap = step >> 3;
      const int cb  = (step & 7) * 32;
      const _Float16* ap = hs + (size_t)(m * KS + tap) * HPITCH;
      v16h af = cat16(*(const v8h*)(ap + cb + 8 * khalf),
                      *(const v8h*)(ap + cb + 16 + 8 * khalf));
      acc = __builtin_amdgcn_wmma_f32_16x16x32_f16(
          false, af, false, bf, (short)0, acc, false, false);
    }

    const float bias = b2[ng];
    #pragma unroll
    for (int v = 0; v < 8; ++v) {
      int mm = v + 8 * khalf;
      out[(size_t)(sb + mm) * ODIM + ng] = acc[v] + bias;
    }
  }
}

// ---------------------------------------------------------------------------
extern "C" void kernel_launch(void* const* d_in, const int* in_sizes, int n_in,
                              void* d_out, int out_size, void* d_ws, size_t ws_size,
                              hipStream_t stream) {
  const int*   node_ids  = (const int*)d_in[0];
  const int*   neighbors = (const int*)d_in[1];
  const float* feat      = (const float*)d_in[2];
  const float* W1        = (const float*)d_in[3];
  const float* b1        = (const float*)d_in[4];
  const float* W2        = (const float*)d_in[5];
  const float* b2        = (const float*)d_in[6];
  float* out = (float*)d_out;

  _Float16* w1t = (_Float16*)d_ws;                 // 256*1280 f16 = 640 KB
  _Float16* w2t = w1t + (size_t)HDIM * KDIM;       // 128*1280 f16 = 320 KB

  const int prep_total = HDIM * KDIM + ODIM * KDIM;  // 491520
  lgc_prep_weights<<<(prep_total + 255) / 256, 256, 0, stream>>>(W1, W2, w1t, w2t);

  lgc_main<<<B_TOTAL / SPB, 256, SMEM_BYTES, stream>>>(
      node_ids, neighbors, feat, w1t, b1, w2t, b2, out);
}
